// KKL_55207509623114
// MI455X (gfx1250) — matrix operand order repaired and checked by reading, not compiled, not run on hardware
//
#include <hip/hip_runtime.h>

typedef __attribute__((ext_vector_type(16))) __bf16 v16bf;
typedef __attribute__((ext_vector_type(8)))  float  v8f;

#define LATENT 64
#define HIDDEN 512
#define NBATCH 1024
#define TOBS   128
#define HOR    160
#define MTILE  32

union Frag { uint4 u[2]; v16bf v; };

__device__ __forceinline__ unsigned short f2bf(float f) {
    union { __bf16 h; unsigned short b; } u; u.h = (__bf16)f; return u.b;
}
__device__ __forceinline__ float bf2f(unsigned short s) {
    union { __bf16 h; unsigned short b; } u; u.b = s; return (float)u.h;
}

// CDNA5 async global->LDS copy (ASYNCcnt). VDST = VGPR with LDS byte offset
// (low 32 bits of the generic LDS address), VADDR = 64-bit global address.
__device__ __forceinline__ void async_ld_b128(const void* g, void* lds_generic) {
    unsigned int lds_off = (unsigned int)(unsigned long long)lds_generic;
    asm volatile("global_load_async_to_lds_b128 %0, %1, off"
                 :: "v"(lds_off), "v"(g) : "memory");
}
__device__ __forceinline__ void wait_async0() {
    asm volatile("s_wait_asynccnt 0" ::: "memory");
}

// A-fragment (16x32 bf16, MxK) from a row-major bf16 LDS buffer.
// ISA 7.12.2: lanes 0-15 = rows M with K pairs {0..7,16..23};
// lanes 16-31 = rows M with K pairs {8..15,24..31}.
__device__ __forceinline__ v16bf load_a(const unsigned short* base, int stride,
                                        int rt, int kt, int lane) {
    int m = rt * 16 + (lane & 15);
    const unsigned short* p = base + m * stride + kt * 32 + ((lane & 16) ? 8 : 0);
    Frag f;
    f.u[0] = *(const uint4*)(p);
    f.u[1] = *(const uint4*)(p + 16);
    return f.v;
}

// B-fragment (32x16 bf16, KxN) from pre-packed tiles: (kt,nt) tile = 32 lanes x 8 dwords.
__device__ __forceinline__ v16bf load_b(const unsigned int* wp, int NT,
                                        int kt, int nt, int lane) {
    const unsigned int* p = wp + (((kt * NT + nt) << 8) + (lane << 3));
    Frag f;
    f.u[0] = *(const uint4*)(p);
    f.u[1] = *(const uint4*)(p + 4);
    return f.v;
}

__device__ __forceinline__ v8f wmma_bf16(v16bf a, v16bf b, v8f c) {
    return __builtin_amdgcn_wmma_f32_16x16x32_bf16(
        false, a, false, b, (short)0, c, false, false);
}

// out = relu(in @ W + bias): M=32 rows, N=HIDDEN, K=KD.
// kt-outer loop: A-fragments loaded once per kt, shared by all 4 N-tiles;
// each B fragment reused by both row tiles. 8 live accumulators.
template <int KD, bool PREF>
__device__ __forceinline__ void layer(const unsigned short* __restrict__ inb,
                                      unsigned short* __restrict__ outb,
                                      const unsigned int* __restrict__ Wp,
                                      const float* __restrict__ bias,
                                      int wave, int lane) {
    const int NT = HIDDEN / 16;
    v8f acc[2][4] = {};
    #pragma unroll 4
    for (int kt = 0; kt < KD / 32; ++kt) {
        if (PREF && kt + 4 < KD / 32)  // global_prefetch_b8 into WGP$
            __builtin_prefetch(Wp + ((((kt + 4) * NT + (wave << 2)) << 8) + (lane << 3)), 0, 0);
        v16bf a0 = load_a(inb, KD, 0, kt, lane);
        v16bf a1 = load_a(inb, KD, 1, kt, lane);
        #pragma unroll
        for (int j = 0; j < 4; ++j) {
            v16bf bfr = load_b(Wp, NT, kt, (wave << 2) + j, lane);
            acc[0][j] = wmma_bf16(a0, bfr, acc[0][j]);
            acc[1][j] = wmma_bf16(a1, bfr, acc[1][j]);
        }
    }
    // C/D layout: VGPR r -> M = r + (lane>=16 ? 8 : 0), N = lane&15 within tile
    int mb = (lane & 16) ? 8 : 0;
    #pragma unroll
    for (int j = 0; j < 4; ++j) {
        int ncol = (((wave << 2) + j) << 4) + (lane & 15);
        float bv = bias[ncol];
        #pragma unroll
        for (int r = 0; r < 8; ++r) {
            float v0 = acc[0][j][r] + bv; v0 = v0 > 0.f ? v0 : 0.f;
            float v1 = acc[1][j][r] + bv; v1 = v1 > 0.f ? v1 : 0.f;
            outb[(mb + r) * HIDDEN + ncol]      = f2bf(v0);
            outb[(16 + mb + r) * HIDDEN + ncol] = f2bf(v1);
        }
    }
}

// Pack f32 weight [Kd x Nd] (or transpose) into bf16 WMMA-B tiles.
__global__ void pack_bfrag(const float* __restrict__ src, unsigned int* __restrict__ dst,
                           int Kd, int Nd, int trans) {
    int idx = blockIdx.x * 256 + threadIdx.x;
    int total = (Kd * Nd) >> 1;
    if (idx >= total) return;
    int v    = idx & 7;
    int lane = (idx >> 3) & 31;
    int tile = idx >> 8;
    int NT = Nd >> 4;
    int kt = tile / NT, nt = tile - kt * NT;
    int n  = (nt << 4) + (lane & 15);
    int k0 = (kt << 5) + ((lane & 16) ? 8 : 0) + ((v & 4) ? 16 : 0) + ((v & 3) << 1);
    float f0 = trans ? src[n * Kd + k0]     : src[k0 * Nd + n];
    float f1 = trans ? src[n * Kd + k0 + 1] : src[(k0 + 1) * Nd + n];
    unsigned int lo = f2bf(f0), hi = f2bf(f1);
    dst[idx] = lo | (hi << 16);
}

__global__ __launch_bounds__(256) void koopman_rollout(
    const float* __restrict__ y, const float* __restrict__ bvec,
    const float* __restrict__ b1, const float* __restrict__ b2,
    const float* __restrict__ b3, const float* __restrict__ b4,
    const float* __restrict__ W5, const float* __restrict__ b5,
    const unsigned int* __restrict__ W1p, const unsigned int* __restrict__ W2p,
    const unsigned int* __restrict__ W3p, const unsigned int* __restrict__ W4p,
    const unsigned int* __restrict__ ATp,
    float* __restrict__ out) {

    __shared__ __align__(16) unsigned short zb[MTILE * LATENT];      // z, bf16 (4KB)
    __shared__ __align__(16) unsigned short hA[MTILE * HIDDEN];      // ping (32KB)
    __shared__ __align__(16) unsigned short hB[MTILE * HIDDEN];      // pong (32KB)
    __shared__ __align__(16) unsigned int   W1l[LATENT * HIDDEN / 2]; // 64KB, loop-invariant
    __shared__ __align__(16) unsigned int   ATl[LATENT * LATENT / 2]; // 8KB, loop-invariant
    __shared__ float ypart[MTILE * 8];
    __shared__ float yin[MTILE];

    const int tid  = threadIdx.x;
    const int wave = tid >> 5;
    const int lane = tid & 31;
    const int rowbase = blockIdx.x * MTILE;

    // One-time async preload of the per-step-reused small weights into LDS.
    for (int i = tid; i < LATENT * HIDDEN / 8; i += 256)
        async_ld_b128(W1p + i * 4, &W1l[i * 4]);
    for (int i = tid; i < LATENT * LATENT / 8; i += 256)
        async_ld_b128(ATp + i * 4, &ATl[i * 4]);

    for (int i = tid; i < MTILE * LATENT / 2; i += 256)
        ((unsigned int*)zb)[i] = 0u;   // z0 = 0

    wait_async0();
    __syncthreads();

    for (int t = 0; t < HOR; ++t) {
        // psi(z): 4 hidden layers
        layer<LATENT, false>(zb, hA, W1l, b1, wave, lane);
        __syncthreads();
        layer<HIDDEN, true>(hA, hB, W2p, b2, wave, lane);
        __syncthreads();
        layer<HIDDEN, true>(hB, hA, W3p, b3, wave, lane);
        __syncthreads();
        layer<HIDDEN, true>(hA, hB, W4p, b4, wave, lane);
        __syncthreads();

        // layer 5: y_hat = h4 @ W5 + b5  (OBS==1 -> per-row dot product)
        {
            int row = tid >> 3, seg = tid & 7;
            const unsigned short* hp = hB + row * HIDDEN + seg * 64;
            const float* wp = W5 + seg * 64;
            float s = 0.f;
            #pragma unroll 8
            for (int k = 0; k < 64; ++k) s += bf2f(hp[k]) * wp[k];
            ypart[row * 8 + seg] = s;
        }
        __syncthreads();
        if (tid < MTILE) {
            float s = b5[0];
            #pragma unroll
            for (int j = 0; j < 8; ++j) s += ypart[tid * 8 + j];
            int grow = rowbase + tid;
            out[grow * HOR + t] = s;
            yin[tid] = (t < TOBS) ? y[grow * TOBS + t] : s;  // teacher-forced vs closed loop
        }
        __syncthreads();

        // dyn: z <- z @ A^T + yin * bvec   (one 16x16 tile per wave)
        {
            int rt = wave >> 2, nt = wave & 3;
            v8f acc = {};
            #pragma unroll
            for (int kt = 0; kt < LATENT / 32; ++kt) {
                v16bf a  = load_a(zb, LATENT, rt, kt, lane);
                v16bf bb = load_b(ATl, LATENT / 16, kt, nt, lane);
                acc = wmma_bf16(a, bb, acc);
            }
            int ncol = (nt << 4) + (lane & 15);
            float bvn = bvec[ncol];
            int mb = rt * 16 + ((lane & 16) ? 8 : 0);
            float vals[8];
            #pragma unroll
            for (int r = 0; r < 8; ++r) vals[r] = acc[r] + yin[mb + r] * bvn;
            __syncthreads();   // all waves done reading old z
            #pragma unroll
            for (int r = 0; r < 8; ++r) zb[(mb + r) * LATENT + ncol] = f2bf(vals[r]);
        }
        __syncthreads();
    }
}

extern "C" void kernel_launch(void* const* d_in, const int* in_sizes, int n_in,
                              void* d_out, int out_size, void* d_ws, size_t ws_size,
                              hipStream_t stream) {
    (void)in_sizes; (void)n_in; (void)out_size; (void)ws_size;
    const float* y    = (const float*)d_in[0];
    const float* A    = (const float*)d_in[1];
    const float* bvec = (const float*)d_in[2];
    const float* W1   = (const float*)d_in[3];
    const float* b1   = (const float*)d_in[4];
    const float* W2   = (const float*)d_in[5];
    const float* b2   = (const float*)d_in[6];
    const float* W3   = (const float*)d_in[7];
    const float* b3   = (const float*)d_in[8];
    const float* W4   = (const float*)d_in[9];
    const float* b4   = (const float*)d_in[10];
    const float* W5   = (const float*)d_in[11];
    const float* b5   = (const float*)d_in[12];
    float* out = (float*)d_out;

    unsigned int* ws  = (unsigned int*)d_ws;
    unsigned int* W1p = ws;                                   // 16384 dw
    unsigned int* W2p = W1p + (LATENT * HIDDEN / 2);          // 131072 dw
    unsigned int* W3p = W2p + (HIDDEN * HIDDEN / 2);
    unsigned int* W4p = W3p + (HIDDEN * HIDDEN / 2);
    unsigned int* ATp = W4p + (HIDDEN * HIDDEN / 2);          // 2048 dw

    hipLaunchKernelGGL(pack_bfrag, dim3((LATENT * HIDDEN / 2) / 256), dim3(256), 0, stream,
                       W1, W1p, LATENT, HIDDEN, 0);
    hipLaunchKernelGGL(pack_bfrag, dim3((HIDDEN * HIDDEN / 2) / 256), dim3(256), 0, stream,
                       W2, W2p, HIDDEN, HIDDEN, 0);
    hipLaunchKernelGGL(pack_bfrag, dim3((HIDDEN * HIDDEN / 2) / 256), dim3(256), 0, stream,
                       W3, W3p, HIDDEN, HIDDEN, 0);
    hipLaunchKernelGGL(pack_bfrag, dim3((HIDDEN * HIDDEN / 2) / 256), dim3(256), 0, stream,
                       W4, W4p, HIDDEN, HIDDEN, 0);
    // B = A^T (B[k][n] = A[n][k]) for the latent-update GEMM
    hipLaunchKernelGGL(pack_bfrag, dim3((LATENT * LATENT / 2) / 256), dim3(256), 0, stream,
                       A, ATp, LATENT, LATENT, 1);

    hipLaunchKernelGGL(koopman_rollout, dim3(NBATCH / MTILE), dim3(256), 0, stream,
                       y, bvec, b1, b2, b3, b4, W5, b5,
                       W1p, W2p, W3p, W4p, ATp, out);
}
